// GAT_75909251990136
// MI455X (gfx1250) — matrix operand order repaired and checked by reading, not compiled, not run on hardware
//
#include <hip/hip_runtime.h>
#include <math.h>

typedef __attribute__((ext_vector_type(16))) __bf16 v16bf;
typedef __attribute__((ext_vector_type(8)))  float  v8f;
typedef __attribute__((ext_vector_type(4)))  unsigned int u32x4;

#define TPB 256
#define NEG_INF_ENC 0x007FFFFFu   // encOrd(-inf)

// ---------- helpers ----------
static __device__ __forceinline__ unsigned short f32_to_bf16(float f) {
  unsigned u = __float_as_uint(f);
  unsigned r = u + 0x7FFFu + ((u >> 16) & 1u);   // round-to-nearest-even
  return (unsigned short)(r >> 16);
}
// order-preserving float<->uint mapping for atomicMax on floats
static __device__ __forceinline__ unsigned encOrd(float f) {
  unsigned u = __float_as_uint(f);
  return (u & 0x80000000u) ? ~u : (u | 0x80000000u);
}
static __device__ __forceinline__ float decOrd(unsigned e) {
  unsigned u = (e & 0x80000000u) ? (e & 0x7FFFFFFFu) : ~e;
  return __uint_as_float(u);
}

// ---------- conversion kernels ----------
__global__ void k_cvt_bf16(const float* __restrict__ in,
                           unsigned short* __restrict__ out, long long n) {
  long long t = (long long)blockIdx.x * blockDim.x + threadIdx.x;
  if (t < n) out[t] = f32_to_bf16(in[t]);
}

// W: K x Nc (row-major f32)  ->  WT: Nc x K (row-major bf16)
__global__ void k_cvtT_bf16(const float* __restrict__ W,
                            unsigned short* __restrict__ WT,
                            int K, int Nc, long long n) {
  long long t = (long long)blockIdx.x * blockDim.x + threadIdx.x;
  if (t >= n) return;
  int o = (int)(t / K);
  int i = (int)(t - (long long)o * K);
  WT[(size_t)o * K + i] = f32_to_bf16(W[(size_t)i * Nc + o]);
}

// ---------- WMMA GEMM:  C[M,Nc] = A[M,K] * B[K,Nc] (+bias)(+relu) ----------
// A row-major bf16; BT = B^T, i.e. BT[n][k], row-major bf16.
// One wave computes one 16x16 tile; K multiple of 32, M,Nc multiples of 16.
__global__ void k_gemm_bf16(const unsigned short* __restrict__ A,
                            const unsigned short* __restrict__ BT,
                            float* __restrict__ C,
                            const float* __restrict__ bias,
                            int M, int K, int Nc, int relu)
{
  int wavesPerBlock = (int)blockDim.x >> 5;
  int waveId = (int)blockIdx.x * wavesPerBlock + ((int)threadIdx.x >> 5);
  int tilesN = Nc >> 4;
  int tilesM = M >> 4;
  int tileM = waveId / tilesN;
  int tileN = waveId - tileM * tilesN;
  if (tileM >= tilesM) return;                  // wave-uniform exit (EXEC all-1 inside)

  int lane = (int)threadIdx.x & 31;
  int half = lane >> 4;                         // 0: lanes 0-15, 1: lanes 16-31
  int lr   = lane & 15;

  const unsigned short* arow = A  + (size_t)(tileM * 16 + lr) * K;  // A row m=lr
  const unsigned short* brow = BT + (size_t)(tileN * 16 + lr) * K;  // B col n=lr

  v8f acc = {};
  for (int k0 = 0; k0 < K; k0 += 32) {
    union { u32x4 q[2]; v16bf v; } af, bfr;
    // A 16x32 bf16 layout: lanes0-15 -> K {0..7 | 16..23}; lanes16-31 -> K {8..15 | 24..31}
    af.q[0] = *(const u32x4*)(arow + k0 + half * 8);
    af.q[1] = *(const u32x4*)(arow + k0 + 16 + half * 8);
    // B 32x16 bf16 layout: lanes0-15 -> K {0..7 | 16..23}; lanes16-31 -> K {8..15 | 24..31}
    bfr.q[0] = *(const u32x4*)(brow + k0 + half * 8);
    bfr.q[1] = *(const u32x4*)(brow + k0 + 16 + half * 8);
    acc = __builtin_amdgcn_wmma_f32_16x16x32_bf16(false, af.v, false, bfr.v,
                                                  (short)0, acc, false, false);
  }
  int on = tileN * 16 + lr;
  float badd = bias ? bias[on] : 0.0f;
#pragma unroll
  for (int r = 0; r < 8; ++r) {                 // C: VGPR r -> row r + 8*half, col lr
    int om = tileM * 16 + half * 8 + r;
    float v = acc[r] + badd;
    if (relu) v = v > 0.0f ? v : 0.0f;
    C[(size_t)om * Nc + on] = v;
  }
}

// ---------- edge-softmax phase ----------
__global__ void k_init_mz(unsigned* __restrict__ mE, float* __restrict__ zE, int n) {
  int t = blockIdx.x * blockDim.x + threadIdx.x;
  if (t < n) { mE[t] = NEG_INF_ENC; zE[t] = 0.0f; }
}

// logit[e,h] = sum_c att[h,c]*leaky_relu(xl[src]+xr[dst]); atomicMax into m[dst,h]
__global__ void k_edge_logit_max(const int* __restrict__ ei, int E, int N, int Hh, int C,
                                 int dout,
                                 const float* __restrict__ xl, const float* __restrict__ xr,
                                 const float* __restrict__ att,
                                 float* __restrict__ logit, unsigned* __restrict__ mE)
{
  long long t = (long long)blockIdx.x * blockDim.x + threadIdx.x;
  long long Et = (long long)E + N;
  if (t >= Et * Hh) return;
  int e = (int)(t / Hh);
  int h = (int)(t - (long long)e * Hh);
  int s, d;
  if (e < E) { s = ei[e]; d = ei[E + e]; } else { s = d = e - E; }
  const float* pl = xl + (size_t)s * dout + h * C;
  const float* pr = xr + (size_t)d * dout + h * C;
  const float* pa = att + h * C;
  float acc = 0.0f;
  for (int c = 0; c < C; ++c) {
    float v = pl[c] + pr[c];
    v = v > 0.0f ? v : 0.2f * v;               // leaky_relu, slope 0.2
    acc += pa[c] * v;
  }
  logit[t] = acc;
  atomicMax(&mE[d * Hh + h], encOrd(acc));
}

// p = exp(logit - m[dst]); z[dst] += p  (logit buffer overwritten with p)
__global__ void k_edge_p_z(const int* __restrict__ ei, int E, int N, int Hh,
                           float* __restrict__ logit,
                           const unsigned* __restrict__ mE, float* __restrict__ zE)
{
  long long t = (long long)blockIdx.x * blockDim.x + threadIdx.x;
  long long Et = (long long)E + N;
  if (t >= Et * Hh) return;
  int e = (int)(t / Hh);
  int h = (int)(t - (long long)e * Hh);
  int d = (e < E) ? ei[E + e] : (e - E);
  float m = decOrd(mE[d * Hh + h]);
  if (!(m > -1e38f)) m = 0.0f;                 // nan_to_num(neginf -> 0)
  float p = expf(logit[t] - m);
  logit[t] = p;
  atomicAdd(&zE[d * Hh + h], p);
}

__global__ void k_init_out(float* __restrict__ out, const float* __restrict__ bias,
                           int dout, long long n) {
  long long t = (long long)blockIdx.x * blockDim.x + threadIdx.x;
  if (t < n) out[t] = bias[(int)(t % dout)];
}

// out[dst, j] += (p/(z+eps)) * xl[src, j]
__global__ void k_edge_agg(const int* __restrict__ ei, int E, int N, int Hh, int C, int dout,
                           const float* __restrict__ p, const float* __restrict__ zE,
                           const float* __restrict__ xl, float* __restrict__ out)
{
  long long t = (long long)blockIdx.x * blockDim.x + threadIdx.x;
  long long Et = (long long)E + N;
  if (t >= Et * dout) return;
  int e = (int)(t / dout);
  int j = (int)(t - (long long)e * dout);
  int s, d;
  if (e < E) { s = ei[e]; d = ei[E + e]; } else { s = d = e - E; }
  int h = j / C;
  float alpha = p[(size_t)e * Hh + h] / (zE[d * Hh + h] + 1e-16f);
  atomicAdd(&out[(size_t)d * dout + j], alpha * xl[(size_t)s * dout + j]);
}

// ---------- GraphNorm + ReLU ----------
__global__ void k_zero_f(float* __restrict__ p, int n) {
  int t = blockIdx.x * blockDim.x + threadIdx.x;
  if (t < n) p[t] = 0.0f;
}
__global__ void k_col_sum(const float* __restrict__ x, int N, int D, float* __restrict__ sum) {
  int j = threadIdx.x;                          // blockDim.x == D
  float acc = 0.0f;
  for (int n = blockIdx.x; n < N; n += gridDim.x) acc += x[(size_t)n * D + j];
  atomicAdd(&sum[j], acc);
}
__global__ void k_col_var(const float* __restrict__ x, int N, int D,
                          const float* __restrict__ sum, const float* __restrict__ gnm,
                          float* __restrict__ sq) {
  int j = threadIdx.x;
  float mean = sum[j] / (float)N;
  float ms = gnm[j];
  float acc = 0.0f;
  for (int n = blockIdx.x; n < N; n += gridDim.x) {
    float xc = x[(size_t)n * D + j] - ms * mean;
    acc += xc * xc;
  }
  atomicAdd(&sq[j], acc);
}
__global__ void k_gn_relu(float* __restrict__ x, int N, int D,
                          const float* __restrict__ sum, const float* __restrict__ sq,
                          const float* __restrict__ gw, const float* __restrict__ gb,
                          const float* __restrict__ gnm, long long n) {
  long long t = (long long)blockIdx.x * blockDim.x + threadIdx.x;
  if (t >= n) return;
  int j = (int)(t % D);
  float mean = sum[j] / (float)N;
  float var  = sq[j] / (float)N;
  float xc = x[t] - gnm[j] * mean;
  float y = gw[j] * xc * rsqrtf(var + 1e-5f) + gb[j];
  x[t] = y > 0.0f ? y : 0.0f;                   // fused ReLU
}

// ---------- pooling ----------
__global__ void k_gate(const float* __restrict__ hid, const float* __restrict__ aW2,
                       const float* __restrict__ ab2, float* __restrict__ gate, int N) {
  int n = blockIdx.x * blockDim.x + threadIdx.x;
  if (n >= N) return;
  const float* h = hid + (size_t)n * 128;
  float acc = ab2[0];
  for (int c = 0; c < 128; ++c) acc += h[c] * aW2[c];
  gate[n] = acc;
}
__global__ void k_pool_init(unsigned* __restrict__ mB, float* __restrict__ zB,
                            float* __restrict__ out, int B, int nout) {
  int t = blockIdx.x * blockDim.x + threadIdx.x;
  if (t < B) { mB[t] = NEG_INF_ENC; zB[t] = 0.0f; }
  if (t < nout) out[t] = 0.0f;
}
__global__ void k_pool_max(const float* __restrict__ gate, const int* __restrict__ batch,
                           unsigned* __restrict__ mB, int N) {
  int n = blockIdx.x * blockDim.x + threadIdx.x;
  if (n < N) atomicMax(&mB[batch[n]], encOrd(gate[n]));
}
__global__ void k_pool_pz(const float* __restrict__ gate, const int* __restrict__ batch,
                          const unsigned* __restrict__ mB, float* __restrict__ zB,
                          float* __restrict__ pp, int N) {
  int n = blockIdx.x * blockDim.x + threadIdx.x;
  if (n >= N) return;
  float m = decOrd(mB[batch[n]]);
  if (!(m > -1e38f)) m = 0.0f;
  float p = expf(gate[n] - m);
  pp[n] = p;
  atomicAdd(&zB[batch[n]], p);
}
__global__ void k_pool_agg(const float* __restrict__ x, const int* __restrict__ batch,
                           const float* __restrict__ pp, const float* __restrict__ zB,
                           float* __restrict__ out, int N) {
  long long t = (long long)blockIdx.x * blockDim.x + threadIdx.x;
  if (t >= (long long)N * 128) return;
  int n = (int)(t >> 7);
  int c = (int)(t & 127);
  int b = batch[n];
  float a = pp[n] / (zB[b] + 1e-16f);
  atomicAdd(&out[(size_t)b * 128 + c], a * x[t]);
}

// ---------- host orchestration ----------
static inline unsigned gridFor(long long n) { return (unsigned)((n + TPB - 1) / TPB); }

extern "C" void kernel_launch(void* const* d_in, const int* in_sizes, int n_in,
                              void* d_out, int out_size, void* d_ws, size_t ws_size,
                              hipStream_t stream)
{
  const int Hh = 4;
  const int N  = in_sizes[0] / 960;
  const int E  = in_sizes[1] / 2;
  const int Bg = out_size / 128;
  const long long Et = (long long)E + N;

  const float* x_in  = (const float*)d_in[0];
  const int*   ei    = (const int*)d_in[1];
  const int*   batch = (const int*)d_in[2];
  const float* aW1 = (const float*)d_in[24];
  const float* ab1 = (const float*)d_in[25];
  const float* aW2 = (const float*)d_in[26];
  const float* ab2 = (const float*)d_in[27];

  const int din_[3] = {960, 512, 256};
  const int C_[3]   = {128, 64, 32};

  // carve workspace
  char* base = (char*)d_ws;
  size_t off = 0;
  auto carve = [&](size_t bytes) -> void* {
    void* p = base + off;
    off += (bytes + 255) & ~(size_t)255;
    return p;
  };
  unsigned short* xbf = (unsigned short*)carve((size_t)N * 960 * 2);
  unsigned short* wlt = (unsigned short*)carve((size_t)512 * 960 * 2);
  unsigned short* wrt = (unsigned short*)carve((size_t)512 * 960 * 2);
  float* xl = (float*)carve((size_t)N * 512 * 4);
  float* xr = (float*)carve((size_t)N * 512 * 4);
  float* xa = (float*)carve((size_t)N * 512 * 4);
  float* xb = (float*)carve((size_t)N * 512 * 4);
  float* lg = (float*)carve((size_t)Et * Hh * 4);
  unsigned* mE = (unsigned*)carve((size_t)N * Hh * 4);
  float* zE = (float*)carve((size_t)N * Hh * 4);
  float* csum = (float*)carve(512 * 4);
  float* csq  = (float*)carve(512 * 4);
  unsigned* mB = (unsigned*)carve((size_t)Bg * 4);
  float* zB = (float*)carve((size_t)Bg * 4);
  (void)ws_size; (void)n_in;

  for (int i = 0; i < 3; ++i) {
    const int din = din_[i], C = C_[i], dout = Hh * C;
    const float* Wl  = (const float*)d_in[3 + 7*i + 0];
    const float* Wr  = (const float*)d_in[3 + 7*i + 1];
    const float* att = (const float*)d_in[3 + 7*i + 2];
    const float* bb  = (const float*)d_in[3 + 7*i + 3];
    const float* gw  = (const float*)d_in[3 + 7*i + 4];
    const float* gb  = (const float*)d_in[3 + 7*i + 5];
    const float* gm  = (const float*)d_in[3 + 7*i + 6];
    const float* cur = (i == 0) ? x_in : ((i == 1) ? xa : xb);
    float* outb = (i == 1) ? xb : xa;

    // bf16 copies of activations and transposed weights
    long long nx = (long long)N * din;
    k_cvt_bf16<<<gridFor(nx), TPB, 0, stream>>>(cur, xbf, nx);
    long long nw = (long long)din * dout;
    k_cvtT_bf16<<<gridFor(nw), TPB, 0, stream>>>(Wl, wlt, din, dout, nw);
    k_cvtT_bf16<<<gridFor(nw), TPB, 0, stream>>>(Wr, wrt, din, dout, nw);

    // XL = x@Wl, XR = x@Wr   (WMMA bf16 -> f32)
    int tiles = (N / 16) * (dout / 16);
    unsigned gb_ = (unsigned)((tiles + 7) / 8);          // 8 waves / 256-thread block
    k_gemm_bf16<<<gb_, 256, 0, stream>>>(xbf, wlt, xl, nullptr, N, din, dout, 0);
    k_gemm_bf16<<<gb_, 256, 0, stream>>>(xbf, wrt, xr, nullptr, N, din, dout, 0);

    // edge softmax over dst
    k_init_mz<<<gridFor(N * Hh), TPB, 0, stream>>>(mE, zE, N * Hh);
    long long nl = Et * Hh;
    k_edge_logit_max<<<gridFor(nl), TPB, 0, stream>>>(ei, E, N, Hh, C, dout, xl, xr, att, lg, mE);
    k_edge_p_z<<<gridFor(nl), TPB, 0, stream>>>(ei, E, N, Hh, lg, mE, zE);

    long long no = (long long)N * dout;
    k_init_out<<<gridFor(no), TPB, 0, stream>>>(outb, bb, dout, no);
    long long ne = Et * dout;
    k_edge_agg<<<gridFor(ne), TPB, 0, stream>>>(ei, E, N, Hh, C, dout, lg, zE, xl, outb);

    // GraphNorm + ReLU (in place on outb)
    k_zero_f<<<gridFor(dout), TPB, 0, stream>>>(csum, dout);
    k_zero_f<<<gridFor(dout), TPB, 0, stream>>>(csq, dout);
    k_col_sum<<<64, dout, 0, stream>>>(outb, N, dout, csum);
    k_col_var<<<64, dout, 0, stream>>>(outb, N, dout, csum, gm, csq);
    k_gn_relu<<<gridFor(no), TPB, 0, stream>>>(outb, N, dout, csum, csq, gw, gb, gm, no);
  }

  // ---------- pooling: gate = relu(x@aW1+ab1)@aW2+ab2, segment softmax over batch ----------
  float* x3 = xa;                                         // layer-3 output
  long long nx = (long long)N * 128;
  k_cvt_bf16<<<gridFor(nx), TPB, 0, stream>>>(x3, xbf, nx);
  k_cvtT_bf16<<<gridFor(128 * 128), TPB, 0, stream>>>(aW1, wlt, 128, 128, 128 * 128);
  int tiles = (N / 16) * (128 / 16);
  k_gemm_bf16<<<(unsigned)((tiles + 7) / 8), 256, 0, stream>>>(xbf, wlt, xl, ab1, N, 128, 128, 1);

  float* gate = xr;
  float* pp   = xr + N;
  k_gate<<<gridFor(N), TPB, 0, stream>>>(xl, aW2, ab2, gate, N);
  k_pool_init<<<gridFor(Bg * 128), TPB, 0, stream>>>(mB, zB, (float*)d_out, Bg, Bg * 128);
  k_pool_max<<<gridFor(N), TPB, 0, stream>>>(gate, batch, mB, N);
  k_pool_pz<<<gridFor(N), TPB, 0, stream>>>(gate, batch, mB, zB, pp, N);
  k_pool_agg<<<gridFor(nx), TPB, 0, stream>>>(x3, batch, pp, zB, (float*)d_out, N);
}
